// MultiHeadAttention_72730976190628
// MI455X (gfx1250) — compile-verified, matrix operand order checked by reading, not run on hardware
//
#include <hip/hip_runtime.h>
#include <hip/hip_bf16.h>

// ---------------------------------------------------------------------------
// MHA for MI455X (gfx1250): bf16 WMMA (V_WMMA_F32_16X16X32_BF16, f32 accum)
//   k1: QKV projection GEMM (fp32 -> bf16; V stored transposed [B,H,DK,S])
//   k2: flash attention, double-buffered GLOBAL_LOAD_ASYNC_TO_LDS_B128 tiles
//       (ASYNCcnt in-order completion -> s_wait_asynccnt 8 drains current tile)
//   k3: output projection GEMM (bf16 A tile async-staged; fp32 d_out)
// ---------------------------------------------------------------------------

#define kB  2
#define kS  2048
#define kD  1024
#define kH  16
#define kDK 64

typedef __attribute__((ext_vector_type(16))) __bf16 bf16x16;
typedef __attribute__((ext_vector_type(8)))  __bf16 bf16x8;
typedef __attribute__((ext_vector_type(4)))  __bf16 bf16x4;
typedef __attribute__((ext_vector_type(8)))  float  f32x8;
typedef __attribute__((ext_vector_type(4)))  float  f32x4;

union FragU { bf16x16 v; bf16x8 h[2]; };

static __device__ __forceinline__ bf16x16 make_frag(bf16x8 lo, bf16x8 hi) {
  FragU f; f.h[0] = lo; f.h[1] = hi; return f.v;
}
static __device__ __forceinline__ __bf16 f2bf(float x) { return (__bf16)x; }

// pack-convert a float4 to 4 contiguous bf16 (one ds_store_b64)
static __device__ __forceinline__ void store_bf4(__bf16* dst, f32x4 v) {
  bf16x4 o;
  o[0] = f2bf(v[0]); o[1] = f2bf(v[1]); o[2] = f2bf(v[2]); o[3] = f2bf(v[3]);
  *(bf16x4*)dst = o;
}

static __device__ __forceinline__ f32x8 wmma_bf16(bf16x16 a, bf16x16 b, f32x8 c) {
  return __builtin_amdgcn_wmma_f32_16x16x32_bf16(
      /*neg_a=*/false, a, /*neg_b=*/false, b,
      /*c_mod=*/(short)0, c, /*reuse_a=*/false, /*reuse_b=*/false);
}

// Async global->LDS 16B copy (CDNA5 GLOBAL_LOAD_ASYNC_TO_LDS_B128, GV mode).
// Low 32 bits of a flat shared-memory address are the wave-relative LDS byte
// address (ISA 10.2 aperture mapping), which is what VDST expects.
static __device__ __forceinline__ void async_copy_b128(void* lds_dst, const void* gsrc) {
  unsigned lds = (unsigned)(uintptr_t)lds_dst;
  asm volatile("global_load_async_to_lds_b128 %0, %1, off"
               :: "v"(lds), "v"(gsrc) : "memory");
}
static __device__ __forceinline__ void wait_async0() {
  asm volatile("s_wait_asynccnt 0x0" ::: "memory");
}
// leave the newest 8 async ops (next tile) in flight; older ones are done
static __device__ __forceinline__ void wait_async8() {
  asm volatile("s_wait_asynccnt 0x8" ::: "memory");
}

// ---------------------------------------------------------------------------
// Kernel 1: Y = X @ W + bias (bf16 out). z=0: Q [B,H,S,DK]; z=1: K [B,H,S,DK];
// z=2: V stored transposed as [B,H,DK,S] so attention can async-copy V tiles.
// Block: 256 thr (8 waves). Tile: M=128, N=64, K-panel=64.
// ---------------------------------------------------------------------------
__global__ __launch_bounds__(256) void qkv_proj_kernel(
    const float* __restrict__ Xq, const float* __restrict__ Xk, const float* __restrict__ Xv,
    const float* __restrict__ Wq, const float* __restrict__ Wk, const float* __restrict__ Wv,
    const float* __restrict__ bq, const float* __restrict__ bk, const float* __restrict__ bv,
    __bf16* __restrict__ Qo, __bf16* __restrict__ Ko, __bf16* __restrict__ Vo) {
  const int z = blockIdx.z;
  const float* X    = (z == 0) ? Xq : (z == 1) ? Xk : Xv;
  const float* W    = (z == 0) ? Wq : (z == 1) ? Wk : Wv;
  const float* bias = (z == 0) ? bq : (z == 1) ? bk : bv;
  __bf16* Out       = (z == 0) ? Qo : (z == 1) ? Ko : Vo;

  const int m0 = blockIdx.y * 128;
  const int n0 = blockIdx.x * 64;

  __shared__ alignas(16) __bf16 As[128][72];   // [m][k]
  __shared__ alignas(16) __bf16 Bs[64][72];    // [n][k]  (W tile transposed)

  const int tid  = threadIdx.x;
  const int lane = tid & 31;
  const int wave = tid >> 5;       // 0..7 -> M strip
  const int hl   = lane >> 4;      // half of wave
  const int l16  = lane & 15;
  const int kb   = hl * 8;         // K sub-chunk per documented A/B layout

  f32x8 acc[4] = {};

  for (int kt = 0; kt < kD; kt += 64) {
    // ---- load A tile (128x64 f32 -> bf16), one b64 LDS store per float4 ----
    for (int i = 0; i < 8; ++i) {
      int idx = tid + i * 256;            // 2048 float4 total
      int r = idx >> 4, c4 = (idx & 15) * 4;
      f32x4 v = *(const f32x4*)(X + (size_t)(m0 + r) * kD + kt + c4);
      store_bf4(&As[r][c4], v);
    }
    // ---- load B tile (64x64 f32, transpose into [n][k]) ----
    for (int i = 0; i < 4; ++i) {
      int idx = tid + i * 256;            // 1024 float4 total
      int r = idx >> 4, c4 = (idx & 15) * 4;   // r = k row, c4 = n col
      f32x4 v = *(const f32x4*)(W + (size_t)(kt + r) * kD + n0 + c4);
      Bs[c4 + 0][r] = f2bf(v[0]); Bs[c4 + 1][r] = f2bf(v[1]);
      Bs[c4 + 2][r] = f2bf(v[2]); Bs[c4 + 3][r] = f2bf(v[3]);
    }
    __syncthreads();

    for (int ks = 0; ks < 64; ks += 32) {
      bf16x16 af = make_frag(*(const bf16x8*)&As[wave * 16 + l16][ks + kb],
                             *(const bf16x8*)&As[wave * 16 + l16][ks + kb + 16]);
      bf16x16 bfr[4];
      for (int nt = 0; nt < 4; ++nt)
        bfr[nt] = make_frag(*(const bf16x8*)&Bs[nt * 16 + l16][ks + kb],
                            *(const bf16x8*)&Bs[nt * 16 + l16][ks + kb + 16]);
      for (int nt = 0; nt < 4; ++nt)
        acc[nt] = wmma_bf16(af, bfr[nt], acc[nt]);
    }
    __syncthreads();
  }

  // ---- epilogue: +bias, bf16 ----
  for (int nt = 0; nt < 4; ++nt) {
    int n  = n0 + nt * 16 + l16;
    int h  = n >> 6, dk = n & 63;
    float bn = bias[n];
    for (int r = 0; r < 8; ++r) {
      int m = m0 + wave * 16 + hl * 8 + r;
      int b = m >> 11, s = m & (kS - 1);
      float y = acc[nt][r] + bn;
      if (z == 2)  // V transposed: [B,H,DK,S]
        Out[(((size_t)(b * kH + h)) * kDK + dk) * kS + s] = f2bf(y);
      else         // Q/K: [B,H,S,DK]
        Out[(((size_t)(b * kH + h)) * kS + s) * kDK + dk] = f2bf(y);
    }
  }
}

// ---------------------------------------------------------------------------
// Kernel 2: flash attention. Block: 128 thr (4 waves), 64 q-rows per block.
// grid = (S/64, H, B). Q,K bf16 [B,H,S,DK]; V bf16 [B,H,DK,S]; O bf16 [B,S,H,DK].
// K/V tiles double-buffered via GLOBAL_LOAD_ASYNC_TO_LDS_B128 (8 async
// instructions per wave per tile; in-order completion).
// ---------------------------------------------------------------------------
__global__ __launch_bounds__(128) void attn_kernel(
    const __bf16* __restrict__ Q, const __bf16* __restrict__ K,
    const __bf16* __restrict__ V, const int* __restrict__ mask,
    __bf16* __restrict__ O) {
  const float scale = 0.125f;  // 1/sqrt(64)
  const int q0 = blockIdx.x * 64;
  const int h  = blockIdx.y;
  const int b  = blockIdx.z;

  const int tid  = threadIdx.x;
  const int lane = tid & 31;
  const int wave = tid >> 5;    // 0..3 -> 16 q-rows each
  const int hl   = lane >> 4;
  const int l16  = lane & 15;
  const int kb   = hl * 8;

  __shared__ alignas(16) __bf16 Ks[2][64][72];    // [buf][key][dk]
  __shared__ alignas(16) __bf16 Vt[2][64][72];    // [buf][dk][key]
  __shared__ alignas(16) __bf16 Ps[4][16][72];    // per-wave P staging

  const size_t bh = ((size_t)b * kH + h) * kS;
  const __bf16* Vbase = V + ((size_t)(b * kH + h) * kDK) * kS;  // [dk][S]

  // Q fragments held in registers for the whole K loop (2 k-steps of 32)
  bf16x16 qf[2];
  {
    const __bf16* qrow = Q + (bh + q0 + wave * 16 + l16) * kDK;
    qf[0] = make_frag(*(const bf16x8*)(qrow + kb),      *(const bf16x8*)(qrow + kb + 16));
    qf[1] = make_frag(*(const bf16x8*)(qrow + 32 + kb), *(const bf16x8*)(qrow + 32 + kb + 16));
  }

  // stage one 64-key tile (8 async b128 copies per thread/wave)
  auto stage = [&](int t, int bufi) {
    const int k0 = t * 64;
    const __bf16* kbase = K + (bh + k0) * kDK;
    for (int i = 0; i < 4; ++i) {
      int idx = tid + i * 128;            // 512 chunks per tile
      int r = idx >> 3, c8 = (idx & 7) * 8;
      async_copy_b128(&Ks[bufi][r][c8], kbase + (size_t)r * kDK + c8);
      async_copy_b128(&Vt[bufi][r][c8], Vbase + (size_t)r * kS + k0 + c8);
    }
  };

  f32x8 oacc[4] = {};
  float mrow[8], lrow[8];
  for (int r = 0; r < 8; ++r) { mrow[r] = -1e30f; lrow[r] = 0.f; }

  const int T = kS / 64;
  stage(0, 0);

  for (int t = 0; t < T; ++t) {
    const int cur = t & 1;
    const int k0  = t * 64;
    if (t + 1 < T) {           // prefetch next tile into the other buffer
      stage(t + 1, cur ^ 1);
      wait_async8();           // drain current tile's 8 copies (in-order)
    } else {
      wait_async0();
    }
    __syncthreads();

    // ---- S = Q @ K^T (16 x 64 per wave) ----
    f32x8 sacc[4] = {};
    for (int ks = 0; ks < 2; ++ks) {
      bf16x16 bfr[4];
      for (int nt = 0; nt < 4; ++nt)
        bfr[nt] = make_frag(*(const bf16x8*)&Ks[cur][nt * 16 + l16][ks * 32 + kb],
                            *(const bf16x8*)&Ks[cur][nt * 16 + l16][ks * 32 + kb + 16]);
      for (int nt = 0; nt < 4; ++nt)
        sacc[nt] = wmma_bf16(qf[ks], bfr[nt], sacc[nt]);
    }

    // ---- scale + mask ----
    const int qrb = q0 + wave * 16 + hl * 8;
    for (int nt = 0; nt < 4; ++nt) {
      int key = k0 + nt * 16 + l16;
      for (int r = 0; r < 8; ++r) {
        float s = sacc[nt][r] * scale;
        int mk = mask[((size_t)b * kS + (qrb + r)) * kS + key];
        sacc[nt][r] = (mk == 0) ? -1e30f : s;
      }
    }

    // ---- online softmax (row stats replicated across 16-lane half) ----
    float scl[8];
    for (int r = 0; r < 8; ++r) {
      float v = fmaxf(fmaxf(sacc[0][r], sacc[1][r]), fmaxf(sacc[2][r], sacc[3][r]));
      for (int off = 1; off < 16; off <<= 1) v = fmaxf(v, __shfl_xor(v, off, 32));
      float mnew = fmaxf(mrow[r], v);
      scl[r] = __expf(mrow[r] - mnew);
      mrow[r] = mnew;
      float rs = 0.f;
      for (int nt = 0; nt < 4; ++nt) {
        float p = __expf(sacc[nt][r] - mnew);
        sacc[nt][r] = p;
        rs += p;
      }
      for (int off = 1; off < 16; off <<= 1) rs += __shfl_xor(rs, off, 32);
      lrow[r] = lrow[r] * scl[r] + rs;
    }

    // ---- rescale O, stage P into A-fragment layout via LDS ----
    for (int nt = 0; nt < 4; ++nt)
      for (int r = 0; r < 8; ++r) {
        oacc[nt][r] *= scl[r];
        Ps[wave][hl * 8 + r][nt * 16 + l16] = f2bf(sacc[nt][r]);
      }

    // ---- O += P @ V ----
    for (int ks = 0; ks < 2; ++ks) {
      bf16x16 pf = make_frag(*(const bf16x8*)&Ps[wave][l16][ks * 32 + kb],
                             *(const bf16x8*)&Ps[wave][l16][ks * 32 + kb + 16]);
      bf16x16 vfr[4];
      for (int nt = 0; nt < 4; ++nt)
        vfr[nt] = make_frag(*(const bf16x8*)&Vt[cur][nt * 16 + l16][ks * 32 + kb],
                            *(const bf16x8*)&Vt[cur][nt * 16 + l16][ks * 32 + kb + 16]);
      for (int nt = 0; nt < 4; ++nt)
        oacc[nt] = wmma_bf16(pf, vfr[nt], oacc[nt]);
    }
    __syncthreads();
  }

  // ---- epilogue: O / l, store bf16 [B,S,H,DK] ----
  for (int r = 0; r < 8; ++r) {
    int qrow = q0 + wave * 16 + hl * 8 + r;
    float inv = 1.f / lrow[r];
    for (int nt = 0; nt < 4; ++nt)
      O[(((size_t)b * kS + qrow) * kH + h) * kDK + nt * 16 + l16] = f2bf(oacc[nt][r] * inv);
  }
}

// ---------------------------------------------------------------------------
// Kernel 3: out = A(bf16, [B*S, D]) @ Wo + bo  -> f32 d_out
// A tile async-staged (raw bf16 copy); W tile converted/transposed via VALU.
// ---------------------------------------------------------------------------
__global__ __launch_bounds__(256) void out_proj_kernel(
    const __bf16* __restrict__ A, const float* __restrict__ W,
    const float* __restrict__ bias, float* __restrict__ Out) {
  const int m0 = blockIdx.y * 128;
  const int n0 = blockIdx.x * 64;

  __shared__ alignas(16) __bf16 As[128][72];
  __shared__ alignas(16) __bf16 Bs[64][72];

  const int tid  = threadIdx.x;
  const int lane = tid & 31;
  const int wave = tid >> 5;
  const int hl   = lane >> 4;
  const int l16  = lane & 15;
  const int kb   = hl * 8;

  f32x8 acc[4] = {};

  for (int kt = 0; kt < kD; kt += 64) {
    for (int i = 0; i < 4; ++i) {                 // A: 1024 16B-chunks, async
      int idx = tid + i * 256;
      int r = idx >> 3, c8 = (idx & 7) * 8;
      async_copy_b128(&As[r][c8], A + (size_t)(m0 + r) * kD + kt + c8);
    }
    for (int i = 0; i < 4; ++i) {                 // B: transpose f32 W tile
      int idx = tid + i * 256;
      int r = idx >> 4, c4 = (idx & 15) * 4;
      f32x4 v = *(const f32x4*)(W + (size_t)(kt + r) * kD + n0 + c4);
      Bs[c4 + 0][r] = f2bf(v[0]); Bs[c4 + 1][r] = f2bf(v[1]);
      Bs[c4 + 2][r] = f2bf(v[2]); Bs[c4 + 3][r] = f2bf(v[3]);
    }
    wait_async0();
    __syncthreads();

    for (int ks = 0; ks < 64; ks += 32) {
      bf16x16 af = make_frag(*(const bf16x8*)&As[wave * 16 + l16][ks + kb],
                             *(const bf16x8*)&As[wave * 16 + l16][ks + kb + 16]);
      bf16x16 bfr[4];
      for (int nt = 0; nt < 4; ++nt)
        bfr[nt] = make_frag(*(const bf16x8*)&Bs[nt * 16 + l16][ks + kb],
                            *(const bf16x8*)&Bs[nt * 16 + l16][ks + kb + 16]);
      for (int nt = 0; nt < 4; ++nt)
        acc[nt] = wmma_bf16(af, bfr[nt], acc[nt]);
    }
    __syncthreads();
  }

  for (int nt = 0; nt < 4; ++nt) {
    int n = n0 + nt * 16 + l16;
    float bn = bias[n];
    for (int r = 0; r < 8; ++r) {
      int m = m0 + wave * 16 + hl * 8 + r;
      Out[(size_t)m * kD + n] = acc[nt][r] + bn;
    }
  }
}

// ---------------------------------------------------------------------------
extern "C" void kernel_launch(void* const* d_in, const int* in_sizes, int n_in,
                              void* d_out, int out_size, void* d_ws, size_t ws_size,
                              hipStream_t stream) {
  (void)in_sizes; (void)n_in; (void)out_size; (void)ws_size;
  const float* query = (const float*)d_in[0];
  const float* key_  = (const float*)d_in[1];
  const float* value = (const float*)d_in[2];
  const int*   mask  = (const int*)d_in[3];
  const float* Wq = (const float*)d_in[4];
  const float* bq = (const float*)d_in[5];
  const float* Wk = (const float*)d_in[6];
  const float* bk = (const float*)d_in[7];
  const float* Wv = (const float*)d_in[8];
  const float* bv = (const float*)d_in[9];
  const float* Wo = (const float*)d_in[10];
  const float* bo = (const float*)d_in[11];
  float* out = (float*)d_out;

  // workspace layout: 4 bf16 buffers of B*S*D elements each (8 MB each)
  const size_t nElem  = (size_t)kB * kS * kD;
  const size_t nBytes = nElem * sizeof(__bf16);
  __bf16* qws = (__bf16*)d_ws;
  __bf16* kws = (__bf16*)((char*)d_ws + 1 * nBytes);
  __bf16* vws = (__bf16*)((char*)d_ws + 2 * nBytes);   // [B,H,DK,S]
  __bf16* ows = (__bf16*)((char*)d_ws + 3 * nBytes);   // [B,S,H,DK]

  dim3 g1(kD / 64, (kB * kS) / 128, 3);
  qkv_proj_kernel<<<g1, 256, 0, stream>>>(query, key_, value,
                                          Wq, Wk, Wv, bq, bk, bv,
                                          qws, kws, vws);

  dim3 g2(kS / 64, kH, kB);
  attn_kernel<<<g2, 128, 0, stream>>>(qws, kws, vws, mask, ows);

  dim3 g3(kD / 64, (kB * kS) / 128, 1);
  out_proj_kernel<<<g3, 256, 0, stream>>>(ows, Wo, bo, out);
}
